// EquiformerV2_OC20_26886495273485
// MI455X (gfx1250) — compile-verified, matrix operand order checked by reading, not run on hardware
//
#include <hip/hip_runtime.h>
#include <hip/hip_bf16.h>

typedef __attribute__((ext_vector_type(16))) __bf16 v16bf;
typedef __attribute__((ext_vector_type(8)))  float  v8f;

#define NN 2500
#define NE 50000
#define NG 16
#define CDIM 128
#define MM 19
#define LL 25
#define NODE_STRIDE (LL*CDIM)   // 3200
#define AVG_DEG 23.395238876342773f
#define AVG_NODES 77.81317f
#define FRAG_HALVES 512          // one 32x16 B fragment = 32 lanes x 16 bf16

__constant__ int dMIDX[MM] = {0, 1,2,3, 4,5,6,7,8, 10,11,12,13,14, 18,19,20,21,22};
__constant__ int dLOF[LL]  = {0, 1,1,1, 2,2,2,2,2, 3,3,3,3,3,3,3, 4,4,4,4,4,4,4,4,4};

// ---------- bf16 helpers ----------
__device__ inline __bf16 to_bf(float f) {
    unsigned u = __float_as_uint(f);
    u = (u + 0x7FFFu + ((u >> 16) & 1u)) >> 16;
    unsigned short h = (unsigned short)u;
    return __builtin_bit_cast(__bf16, h);
}
__device__ inline float bf2f(__bf16 b) {
    unsigned short h = __builtin_bit_cast(unsigned short, b);
    return __uint_as_float(((unsigned)h) << 16);
}
__device__ inline float silu(float x) { return x / (1.f + __expf(-x)); }

// ---------- WMMA fragment helpers (ISA 7.12.2 layouts, wave32) ----------
// A: 16x32 bf16 tile from row-major bf16 matrix; base already includes k0.
__device__ inline v16bf frag_a_bf(const __bf16* base, int stride, int lane) {
    v16bf a;
    const __bf16* p = base + (lane & 15) * stride + ((lane & 16) ? 8 : 0);
#pragma unroll
    for (int h = 0; h < 16; ++h) a[h] = p[(h < 8) ? h : (h + 8)];
    return a;
}
// B: pre-packed fragment -> one contiguous 32B load per lane.
__device__ inline v16bf frag_b_pk(const __bf16* Wpk, int frag, int lane) {
    return *(const v16bf*)(Wpk + ((size_t)frag << 9) + (lane << 4));
}
__device__ inline v8f wmma_bf(v16bf a, v16bf b, v8f c) {
    return __builtin_amdgcn_wmma_f32_16x16x32_bf16(false, a, false, b,
                                                   (short)0, c, false, false);
}

__device__ inline void atomicMaxFloat(float* addr, float val) {
    unsigned* ua = (unsigned*)addr;
    unsigned old = *ua;
    while (true) {
        if (__uint_as_float(old) >= val) break;
        unsigned prev = atomicCAS(ua, old, __float_as_uint(val));
        if (prev == old) break;
        old = prev;
    }
}

// =================== K0: pack f32 weights into bf16 B-fragment layout ===================
// out[frag][lane][16 halves]; frag = ntile*KC + kc; zero-padded past K.
__global__ __launch_bounds__(256) void pack_b_kernel(
    const float* __restrict__ W, int K, int N, int ntiles, int KC,
    __bf16* __restrict__ out) {
    int idx = blockIdx.x * 256 + threadIdx.x;
    int frag = idx >> 5, lane = idx & 31;
    if (frag >= ntiles * KC) return;
    int nt = frag / KC, kc = frag - nt * KC;
    int n  = nt * 16 + (lane & 15);
    int kb = kc * 32 + ((lane & 16) ? 8 : 0);
    __bf16* o = out + ((size_t)frag << 9) + (lane << 4);
#pragma unroll
    for (int h = 0; h < 16; ++h) {
        int k = kb + ((h < 8) ? h : (h + 8));
        float v = (k < K) ? W[(size_t)k * N + n] : 0.f;
        o[h] = to_bf(v);
    }
}

// =================== K1: edge features (smear -> silu MLP) ===================
__global__ __launch_bounds__(256) void edge_feat_kernel(
    const float* __restrict__ ed, const __bf16* __restrict__ We1pk,
    const __bf16* __restrict__ We2pk, __bf16* __restrict__ efeat) {
    __shared__ __bf16 sme[16 * 608];
    __shared__ __bf16 t1[16 * CDIM];
    const int tile = blockIdx.x, tid = threadIdx.x;
    const float delta = 12.0f / 599.0f;
    const float coeff = -0.5f / ((2.f * delta) * (2.f * delta));
    for (int idx = tid; idx < 16 * 608; idx += 256) {
        int r = idx / 608, kk = idx - r * 608;
        float v = 0.f;
        if (kk < 600) {
            float df = ed[tile * 16 + r] - (float)kk * delta;
            v = __expf(coeff * df * df);
        }
        sme[idx] = to_bf(v);
    }
    __syncthreads();
    const int wave = tid >> 5, lane = tid & 31;
    // stage 1: [16x608] x [608x128], wave w -> N-tile w (KC=19)
    {
        v8f c = {0,0,0,0,0,0,0,0};
        for (int kc = 0; kc < 19; ++kc) {
            v16bf a = frag_a_bf(&sme[kc * 32], 608, lane);
            v16bf b = frag_b_pk(We1pk, wave * 19 + kc, lane);
            c = wmma_bf(a, b, c);
        }
        int n = wave * 16 + (lane & 15), mb = (lane & 16) ? 8 : 0;
#pragma unroll
        for (int i = 0; i < 8; ++i) t1[(mb + i) * CDIM + n] = to_bf(silu(c[i]));
    }
    __syncthreads();
    // stage 2: [16x128] x [128x128] (KC=4)
    {
        v8f c = {0,0,0,0,0,0,0,0};
        for (int kc = 0; kc < 4; ++kc) {
            v16bf a = frag_a_bf(&t1[kc * 32], CDIM, lane);
            v16bf b = frag_b_pk(We2pk, wave * 4 + kc, lane);
            c = wmma_bf(a, b, c);
        }
        int n = wave * 16 + (lane & 15), mb = (lane & 16) ? 8 : 0;
#pragma unroll
        for (int i = 0; i < 8; ++i)
            efeat[(size_t)(tile * 16 + mb + i) * CDIM + n] = to_bf(silu(c[i]));
    }
}

// =================== K2: init node state ===================
__global__ __launch_bounds__(256) void init_x_kernel(
    const int* __restrict__ an, const float* __restrict__ semb, float* __restrict__ x) {
    int idx = blockIdx.x * 256 + threadIdx.x;
    if (idx < NN * NODE_STRIDE) {
        int n = idx / NODE_STRIDE, rc = idx - n * NODE_STRIDE;
        x[idx] = (rc < CDIM) ? semb[an[n] * CDIM + rc] : 0.f;
    }
}

// =================== K3: degree scatter (efeat @ W_deg -> x) ===================
__global__ __launch_bounds__(256) void deg_kernel(
    const __bf16* __restrict__ efeat, const __bf16* __restrict__ Wdegpk,
    const int* __restrict__ ei, float* __restrict__ x) {
    const int tile = blockIdx.x;
    const int wave = threadIdx.x >> 5, lane = threadIdx.x & 31;
    const __bf16* A = efeat + (size_t)tile * 16 * CDIM;
    // hoist the 4 A fragments and the 8 target-node indices
    v16bf a0 = frag_a_bf(A +  0, CDIM, lane);
    v16bf a1 = frag_a_bf(A + 32, CDIM, lane);
    v16bf a2 = frag_a_bf(A + 64, CDIM, lane);
    v16bf a3 = frag_a_bf(A + 96, CDIM, lane);
    const int mb = (lane & 16) ? 8 : 0;
    int tg[8];
#pragma unroll
    for (int i = 0; i < 8; ++i) tg[i] = ei[NE + tile * 16 + mb + i];
    for (int j = wave; j < 152; j += 8) {  // 2432/16 N-tiles
        v8f c = {0,0,0,0,0,0,0,0};
        c = wmma_bf(a0, frag_b_pk(Wdegpk, j * 4 + 0, lane), c);
        c = wmma_bf(a1, frag_b_pk(Wdegpk, j * 4 + 1, lane), c);
        c = wmma_bf(a2, frag_b_pk(Wdegpk, j * 4 + 2, lane), c);
        c = wmma_bf(a3, frag_b_pk(Wdegpk, j * 4 + 3, lane), c);
        int col = j * 16 + (lane & 15);
        int m = col >> 7, ch = col & 127;
#pragma unroll
        for (int i = 0; i < 8; ++i)
            atomicAdd(&x[(size_t)tg[i] * NODE_STRIDE + dMIDX[m] * CDIM + ch],
                      c[i] * (1.0f / AVG_DEG));
    }
}

// =================== K4: rms_norm_sh -> bf16 xn ===================
__global__ __launch_bounds__(128) void rms_kernel(
    const float* __restrict__ x, const float* __restrict__ scale, __bf16* __restrict__ xn) {
    const int n = blockIdx.x, c = threadIdx.x;
    __shared__ float red[128];
    __shared__ float inv[5];
    float part[5] = {0, 0, 0, 0, 0};
    for (int l = 0; l < LL; ++l) {
        float v = x[(size_t)n * NODE_STRIDE + l * CDIM + c];
        part[dLOF[l]] += v * v;
    }
    for (int g = 0; g < 5; ++g) {
        red[c] = part[g];
        __syncthreads();
        for (int s = 64; s > 0; s >>= 1) {
            if (c < s) red[c] += red[c + s];
            __syncthreads();
        }
        if (c == 0) inv[g] = rsqrtf(red[0] / ((2.f * g + 1.f) * 128.f) + 1e-8f);
        __syncthreads();
    }
    for (int l = 0; l < LL; ++l) {
        int g = dLOF[l];
        xn[(size_t)n * NODE_STRIDE + l * CDIM + c] =
            to_bf(x[(size_t)n * NODE_STRIDE + l * CDIM + c] * inv[g] * scale[g * CDIM + c]);
    }
}

// =================== K5: q/k projections ===================
__global__ __launch_bounds__(256) void qk_kernel(
    const __bf16* __restrict__ xn, const __bf16* __restrict__ Wqpk,
    const __bf16* __restrict__ Wkpk, float* __restrict__ q, float* __restrict__ k) {
    const int n0 = blockIdx.x * 16;
    const int wave = threadIdx.x >> 5, lane = threadIdx.x & 31;
    // A: row l=0 of 16 consecutive nodes (clamped), hoisted across all tiles
    int r = n0 + (lane & 15);
    if (r >= NN) r = NN - 1;
    const __bf16* p = xn + (size_t)r * NODE_STRIDE + ((lane & 16) ? 8 : 0);
    v16bf av[4];
#pragma unroll
    for (int kc = 0; kc < 4; ++kc) {
        const __bf16* pk = p + kc * 32;
#pragma unroll
        for (int h = 0; h < 16; ++h) av[kc][h] = pk[(h < 8) ? h : (h + 8)];
    }
    for (int t = wave; t < 64; t += 8) {
        const __bf16* Wpk = (t < 32) ? Wqpk : Wkpk;
        float* O = (t < 32) ? q : k;
        int nt = t & 31;
        v8f c = {0,0,0,0,0,0,0,0};
#pragma unroll
        for (int kc = 0; kc < 4; ++kc)
            c = wmma_bf(av[kc], frag_b_pk(Wpk, nt * 4 + kc, lane), c);
        int col = nt * 16 + (lane & 15), mb = (lane & 16) ? 8 : 0;
#pragma unroll
        for (int i = 0; i < 8; ++i) {
            int node = n0 + mb + i;
            if (node < NN) O[(size_t)node * 512 + col] = c[i];
        }
    }
}

// =================== K6: softmax init / logits / exp-sum / rad ===================
__global__ __launch_bounds__(256) void soft_init_kernel(float* mx, float* sm) {
    int idx = blockIdx.x * 256 + threadIdx.x;
    if (idx < NN * 8) { mx[idx] = -3e38f; sm[idx] = 0.f; }
}

__global__ __launch_bounds__(256) void logits_kernel(
    const float* __restrict__ q, const float* __restrict__ k,
    const float* __restrict__ alpha, const int* __restrict__ ei,
    float* __restrict__ logits, float* __restrict__ mx) {
    int idx = blockIdx.x * 256 + threadIdx.x;
    if (idx >= NE * 8) return;
    int e = idx >> 3, h = idx & 7;
    int s = ei[e], t = ei[NE + e];
    const float* qp = q + (size_t)t * 512 + h * 64;
    const float* kp = k + (size_t)s * 512 + h * 64;
    const float* ap = alpha + h * 64;
    float acc = 0.f;
    for (int a = 0; a < 64; ++a) acc += silu(qp[a] + kp[a]) * ap[a];
    acc *= 0.125f;
    logits[idx] = acc;
    atomicMaxFloat(&mx[t * 8 + h], acc);
}

__global__ __launch_bounds__(256) void expsum_kernel(
    const float* __restrict__ logits, const float* __restrict__ mx,
    const int* __restrict__ ei, float* __restrict__ attne, float* __restrict__ sm) {
    int idx = blockIdx.x * 256 + threadIdx.x;
    if (idx >= NE * 8) return;
    int e = idx >> 3, h = idx & 7;
    int t = ei[NE + e];
    float ex = __expf(logits[idx] - mx[t * 8 + h]);
    attne[idx] = ex;
    atomicAdd(&sm[t * 8 + h], ex);
}

__global__ __launch_bounds__(256) void rad_kernel(
    const __bf16* __restrict__ efeat, const float* __restrict__ Wrad, float* __restrict__ rad) {
    int idx = blockIdx.x * 256 + threadIdx.x;
    if (idx >= NE * MM) return;
    int e = idx / MM, m = idx - e * MM;
    const __bf16* ef = efeat + (size_t)e * CDIM;
    float acc = 0.f;
    for (int c = 0; c < CDIM; ++c) acc += bf2f(ef[c]) * Wrad[c * MM + m];
    rad[idx] = silu(acc);
}

// =================== K7: fused attention message (gather->Wv->attn->Wo->scatter) ===================
__global__ __launch_bounds__(256) void msg_kernel(
    const __bf16* __restrict__ xn, const float* __restrict__ rad,
    const float* __restrict__ attne, const float* __restrict__ sm,
    const int* __restrict__ ei, const __bf16* __restrict__ Wvpk,
    const __bf16* __restrict__ Wopk, float* __restrict__ x) {
    const int e = blockIdx.x;
    const int src = ei[e], tgt = ei[NE + e];
    __shared__ __bf16 xm[32 * CDIM];
    __shared__ __bf16 vb[32 * CDIM];
    const int tid = threadIdx.x;
    for (int idx = tid; idx < 32 * CDIM; idx += 256) {
        int m = idx >> 7, c = idx & 127;
        float v = 0.f;
        if (m < MM)
            v = bf2f(xn[(size_t)src * NODE_STRIDE + dMIDX[m] * CDIM + c]) * rad[(size_t)e * MM + m];
        xm[idx] = to_bf(v);
    }
    __syncthreads();
    const int wave = tid >> 5, lane = tid & 31;
    // GEMM1: v = xm @ Wv, then scale by attention
    for (int t = wave; t < 16; t += 8) {
        int mt = t >> 3, nt = t & 7;
        v8f c = {0,0,0,0,0,0,0,0};
#pragma unroll
        for (int kc = 0; kc < 4; ++kc) {
            v16bf a = frag_a_bf(&xm[mt * 16 * CDIM + kc * 32], CDIM, lane);
            c = wmma_bf(a, frag_b_pk(Wvpk, nt * 4 + kc, lane), c);
        }
        int n = nt * 16 + (lane & 15), h = n >> 4, mb = (lane & 16) ? 8 : 0;
        float at = attne[e * 8 + h] / (sm[tgt * 8 + h] + 1e-9f);
#pragma unroll
        for (int i = 0; i < 8; ++i) vb[(mt * 16 + mb + i) * CDIM + n] = to_bf(c[i] * at);
    }
    __syncthreads();
    // GEMM2: out = v @ Wo, scatter-add into x[tgt]
    for (int t = wave; t < 16; t += 8) {
        int mt = t >> 3, nt = t & 7;
        v8f c = {0,0,0,0,0,0,0,0};
#pragma unroll
        for (int kc = 0; kc < 4; ++kc) {
            v16bf a = frag_a_bf(&vb[mt * 16 * CDIM + kc * 32], CDIM, lane);
            c = wmma_bf(a, frag_b_pk(Wopk, nt * 4 + kc, lane), c);
        }
        int n = nt * 16 + (lane & 15), mb = (lane & 16) ? 8 : 0;
#pragma unroll
        for (int i = 0; i < 8; ++i) {
            int m = mt * 16 + mb + i;
            if (m < MM)
                atomicAdd(&x[(size_t)tgt * NODE_STRIDE + dMIDX[m] * CDIM + n], c[i]);
        }
    }
}

// =================== K8: fused FFN with l=0 silu gating ===================
__global__ __launch_bounds__(256) void ffn_kernel(
    const __bf16* __restrict__ xn, const __bf16* __restrict__ W1pk,
    const __bf16* __restrict__ W2pk, float* __restrict__ x) {
    const int n = blockIdx.x, tid = threadIdx.x;
    __shared__ __bf16 xb[32 * CDIM];
    __shared__ float hf[32 * CDIM];
    __shared__ __bf16 hb[32 * CDIM];
    __shared__ float gate[CDIM];
    for (int idx = tid; idx < 32 * CDIM; idx += 256) {
        int m = idx >> 7;
        xb[idx] = (m < LL) ? xn[(size_t)n * NODE_STRIDE + idx] : to_bf(0.f);
    }
    __syncthreads();
    const int wave = tid >> 5, lane = tid & 31;
    for (int t = wave; t < 16; t += 8) {
        int mt = t >> 3, nt = t & 7;
        v8f c = {0,0,0,0,0,0,0,0};
#pragma unroll
        for (int kc = 0; kc < 4; ++kc) {
            v16bf a = frag_a_bf(&xb[mt * 16 * CDIM + kc * 32], CDIM, lane);
            c = wmma_bf(a, frag_b_pk(W1pk, nt * 4 + kc, lane), c);
        }
        int col = nt * 16 + (lane & 15), mb = (lane & 16) ? 8 : 0;
#pragma unroll
        for (int i = 0; i < 8; ++i) hf[(mt * 16 + mb + i) * CDIM + col] = c[i];
    }
    __syncthreads();
    if (tid < CDIM) gate[tid] = silu(hf[tid]);  // silu of l=0 row
    __syncthreads();
    for (int idx = tid; idx < 32 * CDIM; idx += 256)
        hb[idx] = to_bf(hf[idx] * gate[idx & 127]);
    __syncthreads();
    for (int t = wave; t < 16; t += 8) {
        int mt = t >> 3, nt = t & 7;
        v8f c = {0,0,0,0,0,0,0,0};
#pragma unroll
        for (int kc = 0; kc < 4; ++kc) {
            v16bf a = frag_a_bf(&hb[mt * 16 * CDIM + kc * 32], CDIM, lane);
            c = wmma_bf(a, frag_b_pk(W2pk, nt * 4 + kc, lane), c);
        }
        int col = nt * 16 + (lane & 15), mb = (lane & 16) ? 8 : 0;
#pragma unroll
        for (int i = 0; i < 8; ++i) {
            int m = mt * 16 + mb + i;
            if (m < LL) x[(size_t)n * NODE_STRIDE + m * CDIM + col] += c[i];
        }
    }
}

// =================== K9: final energy head (only l=0 matters) ===================
__global__ __launch_bounds__(16) void zero_out_kernel(float* out) {
    if (threadIdx.x < NG) out[threadIdx.x] = 0.f;
}

__global__ __launch_bounds__(128) void energy_kernel(
    const float* __restrict__ x, const float* __restrict__ nsf,
    const float* __restrict__ Wef1, const float* __restrict__ Wef2,
    const int* __restrict__ batch, float* __restrict__ out) {
    const int n = blockIdx.x, c = threadIdx.x;
    __shared__ float red[128];
    __shared__ float xs[128];
    float v = x[(size_t)n * NODE_STRIDE + c];
    red[c] = v * v;
    __syncthreads();
    for (int s = 64; s > 0; s >>= 1) {
        if (c < s) red[c] += red[c + s];
        __syncthreads();
    }
    float inv = rsqrtf(red[0] / 128.f + 1e-8f);
    __syncthreads();
    xs[c] = v * inv * nsf[c];
    __syncthreads();
    float h = 0.f;
    for (int cc = 0; cc < CDIM; ++cc) h += xs[cc] * Wef1[cc * CDIM + c];
    red[c] = h * silu(h) * Wef2[c];
    __syncthreads();
    for (int s = 64; s > 0; s >>= 1) {
        if (c < s) red[c] += red[c + s];
        __syncthreads();
    }
    if (c == 0) atomicAdd(&out[batch[n]], red[0] * (1.0f / AVG_NODES));
}

// ============================== launch ==============================
extern "C" void kernel_launch(void* const* d_in, const int* in_sizes, int n_in,
                              void* d_out, int out_size, void* d_ws, size_t ws_size,
                              hipStream_t stream) {
    const int*   an    = (const int*)d_in[0];
    const int*   ei    = (const int*)d_in[1];
    const float* ed    = (const float*)d_in[2];
    const int*   batch = (const int*)d_in[3];
    const float* semb  = (const float*)d_in[4];
    const float* We1   = (const float*)d_in[5];
    const float* We2   = (const float*)d_in[6];
    const float* Wdeg  = (const float*)d_in[7];
    const float* ns1   = (const float*)d_in[8];
    const float* ns2   = (const float*)d_in[9];
    const float* Wq    = (const float*)d_in[10];
    const float* Wk    = (const float*)d_in[11];
    const float* alpha = (const float*)d_in[12];
    const float* Wv    = (const float*)d_in[13];
    const float* Wrad  = (const float*)d_in[14];
    const float* Wo    = (const float*)d_in[15];
    const float* W1    = (const float*)d_in[16];
    const float* W2    = (const float*)d_in[17];
    const float* nsf   = (const float*)d_in[18];
    const float* Wef1  = (const float*)d_in[19];
    const float* Wef2  = (const float*)d_in[20];
    float* out = (float*)d_out;
    (void)in_sizes; (void)n_in; (void)out_size; (void)ws_size;

    char* p = (char*)d_ws;
    size_t off = 0;
    auto alloc = [&](size_t bytes) -> void* {
        void* r = p + off;
        off = (off + bytes + 255) & ~(size_t)255;
        return r;
    };
    __bf16* efeat  = (__bf16*)alloc((size_t)NE * CDIM * 2);
    float*  x      = (float*) alloc((size_t)NN * NODE_STRIDE * 4);
    __bf16* xn     = (__bf16*)alloc((size_t)NN * NODE_STRIDE * 2);
    float*  q      = (float*) alloc((size_t)NN * 512 * 4);
    float*  k      = (float*) alloc((size_t)NN * 512 * 4);
    float*  logits = (float*) alloc((size_t)NE * 8 * 4);
    float*  attne  = (float*) alloc((size_t)NE * 8 * 4);
    float*  mx     = (float*) alloc((size_t)NN * 8 * 4);
    float*  smx    = (float*) alloc((size_t)NN * 8 * 4);
    float*  rad    = (float*) alloc((size_t)NE * MM * 4);
    // packed weights (bf16 fragment layout)
    __bf16* We1pk  = (__bf16*)alloc((size_t)8 * 19 * FRAG_HALVES * 2);   // 608x128
    __bf16* We2pk  = (__bf16*)alloc((size_t)8 * 4  * FRAG_HALVES * 2);   // 128x128
    __bf16* Wdegpk = (__bf16*)alloc((size_t)152 * 4 * FRAG_HALVES * 2);  // 128x2432
    __bf16* Wqkpk  = (__bf16*)alloc((size_t)4 * 2 * 32 * 4 * FRAG_HALVES * 2); // per layer q,k 128x512
    __bf16* Wsmpk  = (__bf16*)alloc((size_t)4 * 4 * 8 * 4 * FRAG_HALVES * 2);  // per layer Wv,Wo,W1,W2 128x128

    const int F128 = 8 * 4 * FRAG_HALVES;    // halves per packed 128x128
    const int F512 = 32 * 4 * FRAG_HALVES;   // halves per packed 128x512

    // ---- pack weights ----
    pack_b_kernel<<<(8 * 19 * 32 + 255) / 256, 256, 0, stream>>>(We1, 600, CDIM, 8, 19, We1pk);
    pack_b_kernel<<<(8 * 4 * 32 + 255) / 256, 256, 0, stream>>>(We2, CDIM, CDIM, 8, 4, We2pk);
    pack_b_kernel<<<(152 * 4 * 32 + 255) / 256, 256, 0, stream>>>(Wdeg, CDIM, MM * CDIM, 152, 4, Wdegpk);
    for (int i = 0; i < 4; ++i) {
        pack_b_kernel<<<(32 * 4 * 32 + 255) / 256, 256, 0, stream>>>(
            Wq + i * CDIM * 512, CDIM, 512, 32, 4, Wqkpk + (size_t)i * 2 * F512);
        pack_b_kernel<<<(32 * 4 * 32 + 255) / 256, 256, 0, stream>>>(
            Wk + i * CDIM * 512, CDIM, 512, 32, 4, Wqkpk + (size_t)i * 2 * F512 + F512);
        pack_b_kernel<<<(8 * 4 * 32 + 255) / 256, 256, 0, stream>>>(
            Wv + i * CDIM * CDIM, CDIM, CDIM, 8, 4, Wsmpk + (size_t)i * 4 * F128);
        pack_b_kernel<<<(8 * 4 * 32 + 255) / 256, 256, 0, stream>>>(
            Wo + i * CDIM * CDIM, CDIM, CDIM, 8, 4, Wsmpk + (size_t)i * 4 * F128 + F128);
        pack_b_kernel<<<(8 * 4 * 32 + 255) / 256, 256, 0, stream>>>(
            W1 + i * CDIM * CDIM, CDIM, CDIM, 8, 4, Wsmpk + (size_t)i * 4 * F128 + 2 * F128);
        pack_b_kernel<<<(8 * 4 * 32 + 255) / 256, 256, 0, stream>>>(
            W2 + i * CDIM * CDIM, CDIM, CDIM, 8, 4, Wsmpk + (size_t)i * 4 * F128 + 3 * F128);
    }

    // ---- forward ----
    edge_feat_kernel<<<NE / 16, 256, 0, stream>>>(ed, We1pk, We2pk, efeat);
    init_x_kernel<<<(NN * NODE_STRIDE + 255) / 256, 256, 0, stream>>>(an, semb, x);
    deg_kernel<<<NE / 16, 256, 0, stream>>>(efeat, Wdegpk, ei, x);

    for (int i = 0; i < 4; ++i) {
        const __bf16* Wqpk_i = Wqkpk + (size_t)i * 2 * F512;
        const __bf16* Wkpk_i = Wqpk_i + F512;
        const __bf16* Wvpk_i = Wsmpk + (size_t)i * 4 * F128;
        const __bf16* Wopk_i = Wvpk_i + F128;
        const __bf16* W1pk_i = Wvpk_i + 2 * F128;
        const __bf16* W2pk_i = Wvpk_i + 3 * F128;

        rms_kernel<<<NN, 128, 0, stream>>>(x, ns1 + i * 5 * CDIM, xn);
        qk_kernel<<<(NN + 15) / 16, 256, 0, stream>>>(xn, Wqpk_i, Wkpk_i, q, k);
        soft_init_kernel<<<(NN * 8 + 255) / 256, 256, 0, stream>>>(mx, smx);
        logits_kernel<<<(NE * 8 + 255) / 256, 256, 0, stream>>>(q, k, alpha + i * 512, ei,
                                                                logits, mx);
        expsum_kernel<<<(NE * 8 + 255) / 256, 256, 0, stream>>>(logits, mx, ei, attne, smx);
        rad_kernel<<<(NE * MM + 255) / 256, 256, 0, stream>>>(efeat, Wrad + i * CDIM * MM, rad);
        msg_kernel<<<NE, 256, 0, stream>>>(xn, rad, attne, smx, ei, Wvpk_i, Wopk_i, x);
        rms_kernel<<<NN, 128, 0, stream>>>(x, ns2 + i * 5 * CDIM, xn);
        ffn_kernel<<<NN, 256, 0, stream>>>(xn, W1pk_i, W2pk_i, x);
    }

    zero_out_kernel<<<1, 16, 0, stream>>>(out);
    energy_kernel<<<NN, 128, 0, stream>>>(x, nsf, Wef1, Wef2, batch, out);
}
